// RopeAttentionHead_72602127172266
// MI455X (gfx1250) — matrix operand order repaired and checked by reading, not compiled, not run on hardware
//
#include <hip/hip_runtime.h>

// ---------------------------------------------------------------------------
// RoPE attention head for MI455X (gfx1250, wave32, WMMA).
//   x:(8,2048,512) f32, Wq/Wk/Wv:(512,512) f32, R:(2048,512,512) f32
//   out:(8,2048,512) f32
// Pipeline:
//   0) convert_inputs: x, Wq/Wk/Wv -> f16 (one cheap bandwidth pass)
//   1) rope_extract  : gather cos/sin diagonals from R (4 MB instead of 2 GB)
//   2) qkv_rope      : pure-f16 WMMA projections + in-register RoPE,
//                      V stored transposed for the PV matmul
//   3) attn          : causal flash attention, WMMA for S=QK^T and O=PV
// All WMMA B-operand streams are software-pipelined with a 4-slot ring
// (prefetch distance 3) so global-load latency hides behind the WMMA chain.
// ---------------------------------------------------------------------------

#define D_MODEL   512
#define SEQ_LEN   2048
#define BATCH_N   8
#define SCALE_Q   0.04419417382415922f   // 1/sqrt(512), folded into Q

#define NX  ((size_t)BATCH_N * SEQ_LEN * D_MODEL)   // 8388608 x elements
#define NW  ((size_t)D_MODEL * D_MODEL)             // 262144 per weight

typedef __attribute__((ext_vector_type(16))) _Float16 v16h;
typedef __attribute__((ext_vector_type(8)))  _Float16 v8h;
typedef __attribute__((ext_vector_type(8)))  float    v8f;

// --- helpers ---------------------------------------------------------------

__device__ __forceinline__ v8h cvt8(const float* __restrict__ p) {
  float4 a = *(const float4*)p;
  float4 b = *(const float4*)(p + 4);
  v8h r;
  r[0]=(_Float16)a.x; r[1]=(_Float16)a.y; r[2]=(_Float16)a.z; r[3]=(_Float16)a.w;
  r[4]=(_Float16)b.x; r[5]=(_Float16)b.y; r[6]=(_Float16)b.z; r[7]=(_Float16)b.w;
  return r;
}

__device__ __forceinline__ v16h cat16(v8h lo, v8h hi) {
  v16h r;
#pragma unroll
  for (int i = 0; i < 8; ++i) { r[i] = lo[i]; r[i + 8] = hi[i]; }
  return r;
}

__device__ __forceinline__ v8f wmma16(v16h a, v16h b, v8f c) {
  // D = A(16x32 f16) * B(32x16 f16) + C(16x16 f32)
  return __builtin_amdgcn_wmma_f32_16x16x32_f16(false, a, false, b, (short)0, c,
                                                false, false);
}

// A-operand gather (16-bit A 16x32 layout): lane row = lane%16,
// elements e = K[16*(lane/16)+e]  (one contiguous 16-half run).
// B-operand gather (16-bit B 32x16 layout): lane col = lane%16,
// elements e = K[16*(lane/16)+e]  (one contiguous 16-half run).

// --- kernel 0: one-shot f32 -> f16 conversion of x and weights -------------

__global__ __launch_bounds__(256)
void convert_inputs(const float* __restrict__ x,
                    const float* __restrict__ Wq,
                    const float* __restrict__ Wk,
                    const float* __restrict__ Wv,
                    _Float16* __restrict__ xh,
                    _Float16* __restrict__ Wh) {   // Wq|Wk|Wv concatenated
  size_t i8 = ((size_t)blockIdx.x * 256 + threadIdx.x) * 8;
  if (i8 < NX) {
    *(v8h*)(xh + i8) = cvt8(x + i8);
  } else {
    size_t j = i8 - NX;
    if (j < 3 * NW) {
      const float* src = (j < NW) ? (Wq + j)
                        : (j < 2 * NW) ? (Wk + (j - NW))
                                       : (Wv + (j - 2 * NW));
      *(v8h*)(Wh + j) = cvt8(src);
    }
  }
}

// --- kernel 1: gather cos/sin from the block-diagonal rotary matrix --------

__global__ void rope_extract(const float* __restrict__ R,
                             float* __restrict__ cs) {
  int idx = blockIdx.x * blockDim.x + threadIdx.x;   // 2048*256 threads
  if (idx >= SEQ_LEN * (D_MODEL / 2)) return;
  int m = idx >> 8;            // position
  int i = idx & 255;           // rotation pair index
  size_t base = (size_t)m * D_MODEL * D_MODEL;
  float c = R[base + (size_t)(2 * i)     * D_MODEL + 2 * i];   // cos
  float s = R[base + (size_t)(2 * i + 1) * D_MODEL + 2 * i];   // sin
  cs[m * D_MODEL + 2 * i]     = c;
  cs[m * D_MODEL + 2 * i + 1] = s;
}

// --- kernel 2: QKV projection + RoPE, all via WMMA (pure f16 operands) -----
// One block = one (batch, 16-row) tile. 8 waves split 96 output 16x16 tiles.

__global__ __launch_bounds__(256)
void qkv_rope(const _Float16* __restrict__ xh,
              const _Float16* __restrict__ Wh,
              const float* __restrict__ cs,
              _Float16* __restrict__ Qr,
              _Float16* __restrict__ Kr,
              _Float16* __restrict__ VT) {
  const int b    = blockIdx.x / (SEQ_LEN / 16);
  const int m0   = (blockIdx.x % (SEQ_LEN / 16)) * 16;
  const int lane = threadIdx.x & 31;
  const int w    = threadIdx.x >> 5;
  const int half = lane >> 4;
  const int l16  = lane & 15;

  // A-operand: x tile (16 x 512) f16 held in registers, reused for 12 tiles.
  const _Float16* xrow = xh + ((size_t)b * SEQ_LEN + m0 + l16) * D_MODEL;
  v16h areg[16];
#pragma unroll
  for (int ks = 0; ks < 16; ++ks) {
    int k0 = ks * 32 + 16 * half;
    areg[ks] = cat16(*(const v8h*)(xrow + k0), *(const v8h*)(xrow + k0 + 8));
  }

  for (int t = w; t < 96; t += 8) {
    const int mat = t >> 5;              // 0=Q 1=K 2=V
    const int n0  = (t & 31) * 16;
    const _Float16* wrow = Wh + (size_t)mat * NW
                         + (size_t)(n0 + l16) * D_MODEL + 16 * half;

    // ---- K-loop, B stream pipelined (ring of 4, prefetch distance 3) ----
    v8h blo[4], bhi[4];
#pragma unroll
    for (int i = 0; i < 3; ++i) {
      blo[i] = *(const v8h*)(wrow + i * 32);
      bhi[i] = *(const v8h*)(wrow + i * 32 + 8);
    }
    v8f acc = {};
#pragma unroll
    for (int ks = 0; ks < 16; ++ks) {
      if (ks + 3 < 16) {
        blo[(ks + 3) & 3] = *(const v8h*)(wrow + (ks + 3) * 32);
        bhi[(ks + 3) & 3] = *(const v8h*)(wrow + (ks + 3) * 32 + 8);
      }
      acc = wmma16(areg[ks], cat16(blo[ks & 3], bhi[ks & 3]), acc);
    }

    if (mat < 2) {
      // RoPE: columns striped across lanes; pair (2i,2i+1) = lanes (xor 1).
      const int col = n0 + l16;
      const float sgn = (col & 1) ? -1.0f : 1.0f;
      _Float16* dst = ((mat == 0) ? Qr : Kr) + (size_t)b * SEQ_LEN * D_MODEL;
#pragma unroll
      for (int r = 0; r < 8; ++r) {
        int m = m0 + r + 8 * half;
        float2 csv = *(const float2*)(cs + m * D_MODEL + (col & ~1));
        float own  = acc[r];
        float part = __shfl_xor(own, 1, 32);
        float res  = csv.x * own + sgn * csv.y * part;
        if (mat == 0) res *= SCALE_Q;           // fold 1/sqrt(d) into Q
        dst[(size_t)m * D_MODEL + col] = (_Float16)res;
      }
    } else {
      // V stored transposed: VT[b][e][k]; 8 consecutive k per lane -> b128.
      const int e = n0 + l16;
      v8h pack;
#pragma unroll
      for (int r = 0; r < 8; ++r) pack[r] = (_Float16)acc[r];
      _Float16* vt = VT + ((size_t)b * D_MODEL + e) * SEQ_LEN + m0 + 8 * half;
      *(v8h*)vt = pack;
    }
  }
}

// --- kernel 3: causal flash attention, WMMA for S=QK^T and O=PV ------------
// One block = one (batch, 16-row q tile). 8 waves:
//  - S phase: wave w owns k-subtile [kb*128 + 16w .. +15], full d reduction.
//  - PV phase: wave w owns d-slice [64w .. 64w+63] (4 accum tiles).
// Online softmax state (m,l) kept in LDS; P staged as f16 in LDS.

__global__ __launch_bounds__(256)
void attn(const _Float16* __restrict__ Qr,
          const _Float16* __restrict__ Kr,
          const _Float16* __restrict__ VT,
          float* __restrict__ out) {
  const int b    = blockIdx.x / (SEQ_LEN / 16);
  const int m0   = (blockIdx.x % (SEQ_LEN / 16)) * 16;
  const int lane = threadIdx.x & 31;
  const int w    = threadIdx.x >> 5;
  const int half = lane >> 4;
  const int l16  = lane & 15;

  __shared__ __align__(16) _Float16 Plds[16][128];
  __shared__ float red_max[8][16];
  __shared__ float red_sum[8][16];
  __shared__ float bc_mnew[16];
  __shared__ float bc_alpha[16];
  __shared__ float m_row[16];
  __shared__ float l_row[16];

  if (threadIdx.x < 16) { m_row[threadIdx.x] = -3.0e38f; l_row[threadIdx.x] = 0.0f; }
  __syncthreads();

  // Q tile (16 x 512, scale pre-folded) in registers as A-operand.
  const _Float16* qrow = Qr + ((size_t)b * SEQ_LEN + m0 + l16) * D_MODEL;
  v16h areg[16];
#pragma unroll
  for (int ks = 0; ks < 16; ++ks) {
    int k0 = ks * 32 + 16 * half;
    areg[ks] = cat16(*(const v8h*)(qrow + k0), *(const v8h*)(qrow + k0 + 8));
  }

  v8f acc[4];
#pragma unroll
  for (int t2 = 0; t2 < 4; ++t2) acc[t2] = (v8f){};

  const int kbmax = (m0 + 15) / 128;          // causal: skip blocks past diag
  for (int kb = 0; kb <= kbmax; ++kb) {
    const int n0 = kb * 128 + w * 16;
    // wave-uniform: is any column of this subtile unmasked?
    const bool live = (n0 <= m0 + 15);

    // ---- S = Q . K^T for this wave's 16x16 subtile (pipelined B) -------
    v8f s = {};
    if (live) {
      const _Float16* krow = Kr + ((size_t)b * SEQ_LEN + n0 + l16) * D_MODEL
                           + 16 * half;
      v8h blo[4], bhi[4];
#pragma unroll
      for (int i = 0; i < 3; ++i) {
        blo[i] = *(const v8h*)(krow + i * 32);
        bhi[i] = *(const v8h*)(krow + i * 32 + 8);
      }
#pragma unroll
      for (int ks = 0; ks < 16; ++ks) {
        if (ks + 3 < 16) {
          blo[(ks + 3) & 3] = *(const v8h*)(krow + (ks + 3) * 32);
          bhi[(ks + 3) & 3] = *(const v8h*)(krow + (ks + 3) * 32 + 8);
        }
        s = wmma16(areg[ks], cat16(blo[ks & 3], bhi[ks & 3]), s);
      }
    }

    // ---- causal mask + per-row max (across the 16 lanes of each half) --
    const int kcol = n0 + l16;
    float sv[8];
#pragma unroll
    for (int r = 0; r < 8; ++r) {
      int qi = m0 + r + 8 * half;
      float v = s[r];
      if (!live || kcol > qi) v = -3.0e38f;
      sv[r] = v;
    }
#pragma unroll
    for (int r = 0; r < 8; ++r) {
      float v = sv[r];
      for (int off = 1; off < 16; off <<= 1) v = fmaxf(v, __shfl_xor(v, off, 32));
      if (l16 == 0) red_max[w][r + 8 * half] = v;
    }
    __syncthreads();

    if (threadIdx.x < 16) {
      float mo = m_row[threadIdx.x];
      float mn = mo;
#pragma unroll
      for (int ww = 0; ww < 8; ++ww) mn = fmaxf(mn, red_max[ww][threadIdx.x]);
      bc_mnew[threadIdx.x]  = mn;
      bc_alpha[threadIdx.x] = __expf(mo - mn);
      m_row[threadIdx.x]    = mn;
    }
    __syncthreads();

    // ---- P = exp(S - m_new); row sums; stage P f16 in LDS --------------
#pragma unroll
    for (int r = 0; r < 8; ++r) {
      int M = r + 8 * half;
      float p = __expf(sv[r] - bc_mnew[M]);   // masked entries -> 0
      Plds[M][w * 16 + l16] = (_Float16)p;
      float sum = p;
      for (int off = 1; off < 16; off <<= 1) sum += __shfl_xor(sum, off, 32);
      if (l16 == 0) red_sum[w][M] = sum;
    }
    // rescale O accumulators by alpha
#pragma unroll
    for (int r = 0; r < 8; ++r) {
      float a = bc_alpha[r + 8 * half];
#pragma unroll
      for (int t2 = 0; t2 < 4; ++t2) acc[t2][r] *= a;
    }
    __syncthreads();

    if (threadIdx.x < 16) {
      float ls = bc_alpha[threadIdx.x] * l_row[threadIdx.x];
#pragma unroll
      for (int ww = 0; ww < 8; ++ww) ls += red_sum[ww][threadIdx.x];
      l_row[threadIdx.x] = ls;
    }

    // ---- O += P . V  (wave w: d-slice [64w, 64w+63]) -------------------
    // A-operands (P tiles) hoisted from LDS; B (V^T) stream pipelined.
    v16h pa[4];
#pragma unroll
    for (int ks2 = 0; ks2 < 4; ++ks2) {
      int k0 = ks2 * 32 + 16 * half;
      pa[ks2] = cat16(*(const v8h*)&Plds[l16][k0],
                      *(const v8h*)&Plds[l16][k0 + 8]);
    }
    const _Float16* vbase = VT + ((size_t)b * D_MODEL + w * 64 + l16) * SEQ_LEN
                          + kb * 128 + 16 * half;
    // flattened i = ks2*4 + t2; B offset = t2*16*SEQ + ks2*32
    v8h blo[4], bhi[4];
#pragma unroll
    for (int i = 0; i < 3; ++i) {
      const _Float16* p = vbase + (size_t)(i & 3) * 16 * SEQ_LEN + (i >> 2) * 32;
      blo[i] = *(const v8h*)p;
      bhi[i] = *(const v8h*)(p + 8);
    }
#pragma unroll
    for (int i = 0; i < 16; ++i) {
      if (i + 3 < 16) {
        int j = i + 3;
        const _Float16* p = vbase + (size_t)(j & 3) * 16 * SEQ_LEN + (j >> 2) * 32;
        blo[j & 3] = *(const v8h*)p;
        bhi[j & 3] = *(const v8h*)(p + 8);
      }
      acc[i & 3] = wmma16(pa[i >> 2], cat16(blo[i & 3], bhi[i & 3]), acc[i & 3]);
    }
    __syncthreads();   // protect Plds/red arrays for next block, publish l_row
  }

  // ---- epilogue: O / l -----------------------------------------------
  float* obase = out + (size_t)b * SEQ_LEN * D_MODEL;
#pragma unroll
  for (int r = 0; r < 8; ++r) {
    int M = r + 8 * half;
    float inv = 1.0f / l_row[M];
    size_t row = (size_t)(m0 + M) * D_MODEL;
#pragma unroll
    for (int t2 = 0; t2 < 4; ++t2) {
      int dcol = w * 64 + t2 * 16 + l16;
      obase[row + dcol] = acc[t2][r] * inv;
    }
  }
}

// --- launch ----------------------------------------------------------------

extern "C" void kernel_launch(void* const* d_in, const int* in_sizes, int n_in,
                              void* d_out, int out_size, void* d_ws, size_t ws_size,
                              hipStream_t stream) {
  const float* x  = (const float*)d_in[0];
  const float* Wq = (const float*)d_in[1];
  const float* Wk = (const float*)d_in[2];
  const float* Wv = (const float*)d_in[3];
  const float* R  = (const float*)d_in[4];
  float* out = (float*)d_out;

  // workspace layout (bytes):
  //   Qr 16MB | Kr 16MB | VT 16MB | cs 4MB | xh 16MB | Wh 1.5MB   (~69.5 MB)
  const size_t QKV_BYTES = NX * sizeof(_Float16);
  char* ws = (char*)d_ws;
  _Float16* Qr = (_Float16*)(ws);
  _Float16* Kr = (_Float16*)(ws + QKV_BYTES);
  _Float16* VT = (_Float16*)(ws + 2 * QKV_BYTES);
  float*    cs = (float*)   (ws + 3 * QKV_BYTES);
  _Float16* xh = (_Float16*)(ws + 3 * QKV_BYTES
                                + (size_t)SEQ_LEN * D_MODEL * sizeof(float));
  _Float16* Wh = (_Float16*)((char*)xh + QKV_BYTES);

  const int tiles = BATCH_N * SEQ_LEN / 16;   // 1024 blocks

  const int conv_threads = (int)((NX + 3 * NW) / 8);     // 8 elems per thread
  convert_inputs<<<(conv_threads + 255) / 256, 256, 0, stream>>>(
      x, Wq, Wk, Wv, xh, Wh);
  rope_extract<<<(SEQ_LEN * (D_MODEL / 2)) / 256, 256, 0, stream>>>(R, cs);
  qkv_rope<<<tiles, 256, 0, stream>>>(xh, Wh, cs, Qr, Kr, VT);
  attn<<<tiles, 256, 0, stream>>>(Qr, Kr, VT, out);
}
